// MultiTemporalCrossTransformer_17317308137614
// MI455X (gfx1250) — compile-verified, matrix operand order checked by reading, not run on hardware
//
#include <hip/hip_runtime.h>

// ---------------- types ----------------
typedef __bf16 bf16;
typedef __attribute__((ext_vector_type(16))) __bf16 v16bf;
typedef __attribute__((ext_vector_type(8)))  float  v8f;
typedef __attribute__((ext_vector_type(4)))  int    v4i;
typedef unsigned int u32;

// ---------------- problem constants ----------------
#define NSUP 25
#define NQ   50
#define NALL 75
#define SEQ  12
#define IND  1152
#define DM   1152
#define HEADS 2
#define DH   576
#define TT   220          // C(12,3)
#define WAY  5
#define SHOT 5
#define MPAD 224          // 7 * 32 row blocks for queries
#define KPAD 1152         // 12 * 96 key blocks (1100 real keys)
#define MROWS_PAD 960     // 15 * 64
#define NOUT 6912         // 2 (k,v) * 3 slots * 1152

// ---------------- workspace layout (bytes) ----------------
#define OFF_XB    0ull                      // [960][1152] bf16
#define OFF_KQ    2211840ull                // [50][2][224][576] bf16 (LN'd q keys)
#define OFF_KS    28016640ull               // [5][2][1152][576] bf16 (LN'd s keys)
#define OFF_VST   41287680ull               // [5][2][576][1152] bf16 (s values, d-major)
#define OFF_ACC   54558720ull               // [256] f32 accumulators
#define OFF_WT    54559744ull               // [6912][1152] bf16 (transposed weights)
#define OFF_AOUT  70484992ull               // [960][6912] f32 (frame projections)
#define OFF_QV    97027072ull               // [50][2][224][576] f32 (q values)
#define ZERO_U4   3409984                   // (OFF_WT)/16  -> zero Xb..acc each call

// ---------------- CDNA5 async global->LDS staging ----------------
#ifdef __has_builtin
#if __has_builtin(__builtin_amdgcn_global_load_async_to_lds_b128)
#define HAVE_ASYNC 1
#endif
#endif
#ifndef HAVE_ASYNC
#define HAVE_ASYNC 0
#endif

__device__ __forceinline__ void cp128(bf16* lds_dst, const bf16* gsrc) {
#if HAVE_ASYNC
  __builtin_amdgcn_global_load_async_to_lds_b128((v4i*)gsrc, (v4i*)lds_dst, 0, 0);
#else
  *(uint4*)lds_dst = *(const uint4*)gsrc;
#endif
}

template <int N>
__device__ __forceinline__ void async_waitcnt() {
#if HAVE_ASYNC
#if __has_builtin(__builtin_amdgcn_s_wait_asynccnt)
  __builtin_amdgcn_s_wait_asynccnt(N);
#else
  asm volatile("s_wait_asynccnt %0" ::"i"(N) : "memory");
#endif
#endif
}

// ---------------- WMMA helpers ----------------
__device__ __forceinline__ v8f wmma_bf16(v16bf a, v16bf b, v8f c) {
  return __builtin_amdgcn_wmma_f32_16x16x32_bf16(false, a, false, b, (short)0, c, false, false);
}

// Fragment load from LDS, row-major [outer][inner-K], per CDNA5 16-bit A layout:
// lane<16: outer = outer0+lane, K pairs {0,1},{2,3},{4,5},{6,7},{16,17},... ; lane>=16: +8 K offset.
// Same packing serves B when storage is [N][K].
__device__ __forceinline__ v16bf ldfrag(const bf16* base, int outer0, int stride, int k0, int lane) {
  const bf16* p = base + (outer0 + (lane & 15)) * stride + k0 + ((lane & 16) ? 8 : 0);
  v16bf r;
#pragma unroll
  for (int v = 0; v < 8; ++v) {
    int k = ((v & 3) << 1) + ((v >> 2) << 4);
    r[2 * v]     = p[k];
    r[2 * v + 1] = p[k + 1];
  }
  return r;
}

__device__ __forceinline__ float hmax16(float v) {
#pragma unroll
  for (int m = 8; m >= 1; m >>= 1) v = fmaxf(v, __shfl_xor(v, m, 32));
  return v;
}
__device__ __forceinline__ float hsum16(float v) {
#pragma unroll
  for (int m = 8; m >= 1; m >>= 1) v += __shfl_xor(v, m, 32);
  return v;
}

// ---------------- kernel 0: zero scratch prefix ----------------
__global__ void zero_kernel(uint4* p, int n) {
  int i = blockIdx.x * blockDim.x + threadIdx.x;
  if (i < n) p[i] = make_uint4(0u, 0u, 0u, 0u);
}

// ---------------- kernel 1: X + positional encoding -> bf16 ----------------
__global__ void prep_x_kernel(const float* __restrict__ support, const float* __restrict__ queries,
                              bf16* __restrict__ Xb) {
  int idx = blockIdx.x * blockDim.x + threadIdx.x;
  if (idx >= NALL * SEQ * IND) return;
  int i  = idx % IND;
  int fn = idx / IND;
  int f  = fn % SEQ;
  int n  = fn / SEQ;
  const float* src = (n < NSUP) ? (support + (size_t)n * SEQ * IND)
                                : (queries + (size_t)(n - NSUP) * SEQ * IND);
  double div = exp((double)(i >> 1) * (-2.0 * log(10000.0) / (double)IND));
  double ang = (double)f * div;
  float pe = (float)(((i & 1) ? cos(ang) : sin(ang)) * 0.1);
  Xb[(size_t)fn * IND + i] = (bf16)(src[f * IND + i] + pe);
}

// ---------------- kernel 2: weight transpose/convert ----------------
// Wt[kv*3456 + c*1152 + o][i] = w_kv[c*1152 + i][o]   (bf16)
__global__ void prep_w_kernel(const float* __restrict__ k_w, const float* __restrict__ v_w,
                              bf16* __restrict__ Wt) {
  __shared__ float tile[32][33];
  int z = blockIdx.z, kv = z / 3, c = z % 3;
  const float* src = kv ? v_w : k_w;
  int i0 = blockIdx.x * 32, o0 = blockIdx.y * 32;
  int tx = threadIdx.x, ty = threadIdx.y;      // 32 x 8
#pragma unroll
  for (int r = 0; r < 32; r += 8)
    tile[ty + r][tx] = src[(size_t)(c * IND + i0 + ty + r) * DM + (o0 + tx)];
  __syncthreads();
#pragma unroll
  for (int r = 0; r < 32; r += 8) {
    int o = o0 + ty + r;
    Wt[(size_t)(kv * 3456 + c * IND + o) * IND + (i0 + tx)] = (bf16)tile[tx][ty + r];
  }
}

// ---------------- kernel 3: projection GEMM  A[960x6912] = Xb[960x1152] @ Wt^T ----------------
#define PJ_ST 40
__global__ __launch_bounds__(128) void proj_kernel(const bf16* __restrict__ Xb,
                                                   const bf16* __restrict__ Wt,
                                                   float* __restrict__ Aout) {
  __shared__ bf16 At[64 * PJ_ST];
  __shared__ bf16 Bt[64 * PJ_ST];
  int tid = threadIdx.x, wave = tid >> 5, lane = tid & 31;
  int n0 = blockIdx.x * 64, m0 = blockIdx.y * 64;
  int sr = tid >> 2, sc = tid & 3;             // 32 rows x 4 chunks per pass
  v8f z8 = {0.f, 0.f, 0.f, 0.f, 0.f, 0.f, 0.f, 0.f};
  v8f acc[4] = {z8, z8, z8, z8};
  for (int kc = 0; kc < 36; ++kc) {
    __syncthreads();
#pragma unroll
    for (int rr = 0; rr < 64; rr += 32) {
      cp128(At + (rr + sr) * PJ_ST + sc * 8,
            Xb + (size_t)(m0 + rr + sr) * IND + kc * 32 + sc * 8);
      cp128(Bt + (rr + sr) * PJ_ST + sc * 8,
            Wt + (size_t)(n0 + rr + sr) * IND + kc * 32 + sc * 8);
    }
    async_waitcnt<0>();
    __syncthreads();
    v16bf a = ldfrag(At, wave * 16, PJ_ST, 0, lane);
    v16bf b0 = ldfrag(Bt, 0, PJ_ST, 0, lane);
    v16bf b1 = ldfrag(Bt, 16, PJ_ST, 0, lane);
    v16bf b2 = ldfrag(Bt, 32, PJ_ST, 0, lane);
    v16bf b3 = ldfrag(Bt, 48, PJ_ST, 0, lane);
    acc[0] = wmma_bf16(a, b0, acc[0]);
    acc[1] = wmma_bf16(a, b1, acc[1]);
    acc[2] = wmma_bf16(a, b2, acc[2]);
    acc[3] = wmma_bf16(a, b3, acc[3]);
  }
#pragma unroll
  for (int nt = 0; nt < 4; ++nt)
#pragma unroll
    for (int j = 0; j < 8; ++j) {
      int r = m0 + wave * 16 + j + ((lane & 16) ? 8 : 0);
      int c = n0 + nt * 16 + (lane & 15);
      Aout[(size_t)r * NOUT + c] = acc[nt][j];
    }
}

// ---------------- kernel 4: tuple gather + bias + LN, scatter to attention layouts ----------------
__global__ __launch_bounds__(128) void gather_ln_kernel(const float* __restrict__ Aout,
    const float* __restrict__ k_b, const float* __restrict__ v_b,
    const float* __restrict__ ln_g, const float* __restrict__ ln_b,
    bf16* __restrict__ Kq, bf16* __restrict__ Ks, bf16* __restrict__ Vst,
    float* __restrict__ Qv) {
  int n = blockIdx.x, t = blockIdx.y;
  int tid = threadIdx.x, wave = tid >> 5, lane = tid & 31;
  // decode combination t -> (fa < fb < fc)
  int fa = 0, idx = t;
  while (true) { int cnt = (11 - fa) * (10 - fa) / 2; if (idx < cnt) break; idx -= cnt; ++fa; }
  int fb = fa + 1;
  while (true) { int cnt = 11 - fb; if (idx < cnt) break; idx -= cnt; ++fb; }
  int fc = fb + 1 + idx;
  const float* A0 = Aout + (size_t)(n * SEQ + fa) * NOUT;
  const float* A1 = Aout + (size_t)(n * SEQ + fb) * NOUT + IND;
  const float* A2 = Aout + (size_t)(n * SEQ + fc) * NOUT + 2 * IND;
  float kv[9];
  float lsum = 0.f, lsq = 0.f;
#pragma unroll
  for (int ii = 0; ii < 9; ++ii) {
    int o = tid + ii * 128;
    float v = k_b[o] + A0[o] + A1[o] + A2[o];
    kv[ii] = v; lsum += v; lsq += v * v;
  }
  __shared__ float rbuf[8];
#pragma unroll
  for (int m = 16; m >= 1; m >>= 1) { lsum += __shfl_xor(lsum, m, 32); lsq += __shfl_xor(lsq, m, 32); }
  if (lane == 0) { rbuf[wave] = lsum; rbuf[4 + wave] = lsq; }
  __syncthreads();
  float ts = rbuf[0] + rbuf[1] + rbuf[2] + rbuf[3];
  float tq = rbuf[4] + rbuf[5] + rbuf[6] + rbuf[7];
  float mu = ts * (1.0f / DM);
  float var = tq * (1.0f / DM) - mu * mu;
  float rs = rsqrtf(var + 1e-5f);
  bool sup = (n < NSUP);
  int w = n / SHOT, s = n % SHOT, qn = n - NSUP;
#pragma unroll
  for (int ii = 0; ii < 9; ++ii) {
    int o = tid + ii * 128;
    int h = o / DH, d = o % DH;
    float kval = (kv[ii] - mu) * rs * ln_g[o] + ln_b[o];
    float vval = v_b[o] + A0[3456 + o] + A1[3456 + o] + A2[3456 + o];
    if (sup) {
      int key = s * TT + t;
      Ks [((size_t)(w * HEADS + h) * KPAD + key) * DH + d] = (bf16)kval;
      Vst[((size_t)(w * HEADS + h) * DH + d) * KPAD + key] = (bf16)vval;
    } else {
      Kq[((size_t)(qn * HEADS + h) * MPAD + t) * DH + d] = (bf16)kval;
      Qv[((size_t)(qn * HEADS + h) * MPAD + t) * DH + d] = vval;
    }
  }
}

// ---------------- kernel 5: fused flash attention + prototype distance ----------------
// grid (7 row-blocks, 50 queries, 10 = way*head), 192 threads (6 waves)
// single-pass online softmax; K tile double-buffered, first V tile trickled
// behind the S phase via counted ASYNCcnt waits (async completions are in-order)
#define QS_ST 584
#define KB_ST 72
#define VT_ST 40
#define PS_ST 98
#define KB_HALFS (96 * KB_ST)
#define ATTN_LDS ((32*QS_ST + 2*KB_HALFS + 576*VT_ST + 32*PS_ST) * 2 + (6*32*2 + 3*32) * 4)

__global__ __launch_bounds__(192) void attn_kernel(const bf16* __restrict__ Kq,
    const bf16* __restrict__ Ks, const bf16* __restrict__ Vst,
    const float* __restrict__ Qv, float* __restrict__ acc) {
  extern __shared__ __align__(16) char smem_raw[];
  bf16* Qs  = (bf16*)smem_raw;                 // [32][584]
  bf16* Kb0 = Qs + 32 * QS_ST;                 // 2 x [96][72]
  bf16* Vt  = Kb0 + 2 * KB_HALFS;              // [576][40]
  bf16* Ps  = Vt + 576 * VT_ST;                // [32][98]
  float* stm   = (float*)(Ps + 32 * PS_ST);    // [6][32] row-max partials
  float* stl   = stm + 192;                    // [6][32] row-sum partials
  float* Mrow  = stl + 192;                    // [32] running max
  float* Lrow  = Mrow + 32;                    // [32] running sum
  float* Alpha = Lrow + 32;                    // [32] rescale factor

  int tid = threadIdx.x, wave = tid >> 5, lane = tid & 31;
  int mb = blockIdx.x, q = blockIdx.y, wh = blockIdx.z;
  int w = wh >> 1, h = wh & 1;
  int m0 = mb * 32;
  const bf16* gQ = Kq + ((size_t)(q * HEADS + h) * MPAD + m0) * DH;
  const bf16* gK = Ks + (size_t)(w * HEADS + h) * KPAD * DH;
  const bf16* gV = Vst + (size_t)(w * HEADS + h) * DH * KPAD;

  // stage Q block: 32 rows x 72 chunks; 6 threads per row, 12 chunks each
  {
    int qr = tid & 31, qc0 = tid >> 5;
#pragma unroll
    for (int i = 0; i < 12; ++i) {
      int ch = qc0 + 6 * i;
      cp128(Qs + qr * QS_ST + ch * 8, gQ + qr * DH + ch * 8);
    }
  }
  if (tid < 32) { Mrow[tid] = -1e30f; Lrow[tid] = 0.f; }

  int kk0 = tid >> 3, kch = tid & 7;           // K tile: 96 keys x 8 chunks
  auto stage_kb = [&](int buf, int it, int dc) {
    bf16* kb = Kb0 + buf * KB_HALFS;
#pragma unroll
    for (int i = 0; i < 4; ++i) {
      int kk = kk0 + i * 24;
      cp128(kb + kk * KB_ST + kch * 8, gK + (size_t)(it * 96 + kk) * DH + dc * 64 + kch * 8);
    }
  };
  int vd0 = tid >> 2, vch = tid & 3;           // V tile: 576 d-rows x 4 chunks
  auto stage_vt = [&](int keybase) {
#pragma unroll
    for (int i = 0; i < 12; ++i) {
      int d = vd0 + i * 48;
      cp128(Vt + d * VT_ST + vch * 8, gV + (size_t)d * KPAD + keybase + vch * 8);
    }
  };
  auto stage_vt_piece = [&](int keybase, int p) {  // 2 of the 12 chunks
#pragma unroll
    for (int i = 2 * p; i < 2 * p + 2; ++i) {
      int d = vd0 + i * 48;
      cp128(Vt + d * VT_ST + vch * 8, gV + (size_t)d * KPAD + keybase + vch * 8);
    }
  };

  const float scale = 0.0416666667f;           // 1/sqrt(576)
  v8f z8 = {0.f, 0.f, 0.f, 0.f, 0.f, 0.f, 0.f, 0.f};
  v8f oacc[12];
#pragma unroll
  for (int i = 0; i < 12; ++i) oacc[i] = z8;

  for (int it = 0; it < 12; ++it) {
    // ---- S = Q @ K^T over 96-key tile; double-buffered K, V(kc=0) trickled ----
    v8f st[2] = {z8, z8};
    __syncthreads();                           // prev users of Kb/Ps/Vt done
    stage_kb(0, it, 0);
    for (int dc = 0; dc < 9; ++dc) {
      // counted wait: current K buffer is older than the <=2 newest V-piece ops
      if (dc == 0 || dc >= 7) async_waitcnt<0>(); else async_waitcnt<2>();
      __syncthreads();                         // buf[dc&1] visible to all waves
      if (dc < 8) stage_kb((dc + 1) & 1, it, dc + 1);
      if (dc < 6) stage_vt_piece(it * 96, dc); // prefetch V for kc=0
      const bf16* kb = Kb0 + (dc & 1) * KB_HALFS;
      v16bf bk0  = ldfrag(kb, wave * 16, KB_ST, 0, lane);
      v16bf bk1  = ldfrag(kb, wave * 16, KB_ST, 32, lane);
      v16bf aq00 = ldfrag(Qs, 0,  QS_ST, dc * 64,      lane);
      v16bf aq10 = ldfrag(Qs, 16, QS_ST, dc * 64,      lane);
      v16bf aq01 = ldfrag(Qs, 0,  QS_ST, dc * 64 + 32, lane);
      v16bf aq11 = ldfrag(Qs, 16, QS_ST, dc * 64 + 32, lane);
      st[0] = wmma_bf16(aq00, bk0, st[0]);
      st[1] = wmma_bf16(aq10, bk0, st[1]);
      st[0] = wmma_bf16(aq01, bk1, st[0]);
      st[1] = wmma_bf16(aq11, bk1, st[1]);
    }
    // ---- online softmax update ----
    int colg = it * 96 + wave * 16 + (lane & 15);
    bool cvalid = colg < (SHOT * TT);
    float sv[2][8];
#pragma unroll
    for (int tt = 0; tt < 2; ++tt)
#pragma unroll
      for (int j = 0; j < 8; ++j) {
        float v = cvalid ? st[tt][j] * scale : -1e30f;
        sv[tt][j] = v;
        float rm = hmax16(v);
        if ((lane & 15) == 0) {
          int r = tt * 16 + j + ((lane & 16) ? 8 : 0);
          stm[wave * 32 + r] = rm;
        }
      }
    __syncthreads();
    if (tid < 32) {
      float mi = stm[tid];
      for (int i = 1; i < 6; ++i) mi = fmaxf(mi, stm[i * 32 + tid]);
      float Mo = Mrow[tid], Mn = fmaxf(Mo, mi);
      Alpha[tid] = __expf(Mo - Mn);
      Mrow[tid] = Mn;
    }
    __syncthreads();
    int colL = wave * 16 + (lane & 15);
#pragma unroll
    for (int tt = 0; tt < 2; ++tt)
#pragma unroll
      for (int j = 0; j < 8; ++j) {
        int r = tt * 16 + j + ((lane & 16) ? 8 : 0);
        float p = __expf(sv[tt][j] - Mrow[r]);
        Ps[r * PS_ST + colL] = (bf16)p;
        float ps = hsum16(p);
        if ((lane & 15) == 0) stl[wave * 32 + r] = ps;
        float a = Alpha[r];
#pragma unroll
        for (int nt = 0; nt < 6; ++nt) oacc[tt * 6 + nt][j] *= a;
      }
    async_waitcnt<0>();                        // V(kc=0) fully resident
    __syncthreads();                           // publishes Ps + stl + Vt
    if (tid < 32) {
      float s = 0.f;
      for (int i = 0; i < 6; ++i) s += stl[i * 32 + tid];
      Lrow[tid] = Lrow[tid] * Alpha[tid] + s;
    }
    // ---- O += P @ V ----
    for (int kc = 0; kc < 3; ++kc) {
      if (kc > 0) {
        __syncthreads();                       // prev Vt readers done
        stage_vt(it * 96 + kc * 32);
        async_waitcnt<0>();
        __syncthreads();
      }
      v16bf bv[6];
#pragma unroll
      for (int nt = 0; nt < 6; ++nt)
        bv[nt] = ldfrag(Vt, wave * 96 + nt * 16, VT_ST, 0, lane);
#pragma unroll
      for (int tt = 0; tt < 2; ++tt) {
        v16bf ap = ldfrag(Ps, tt * 16, PS_ST, kc * 32, lane);
#pragma unroll
        for (int nt = 0; nt < 6; ++nt)
          oacc[tt * 6 + nt] = wmma_bf16(ap, bv[nt], oacc[tt * 6 + nt]);
      }
    }
  }
  __syncthreads();

  // ---- epilogue: normalize, diff vs q_vs, squared sum ----
  float lsum = 0.f;
#pragma unroll
  for (int tt = 0; tt < 2; ++tt)
#pragma unroll
    for (int j = 0; j < 8; ++j) {
      int r = tt * 16 + j + ((lane & 16) ? 8 : 0);
      int gr = m0 + r;
      if (gr < TT) {
        float linv = 1.0f / Lrow[r];
#pragma unroll
        for (int nt = 0; nt < 6; ++nt) {
          float ov = oacc[tt * 6 + nt][j] * linv;
          int dcol = wave * 96 + nt * 16 + (lane & 15);
          float qv = Qv[((size_t)(q * HEADS + h) * MPAD + gr) * DH + dcol];
          float d = qv - ov;
          lsum += d * d;
        }
      }
    }
#pragma unroll
  for (int m = 16; m >= 1; m >>= 1) lsum += __shfl_xor(lsum, m, 32);
  __syncthreads();
  if (lane == 0) stm[wave] = lsum;
  __syncthreads();
  if (tid == 0) {
    float s = 0.f;
    for (int i = 0; i < 6; ++i) s += stm[i];
    atomicAdd(&acc[q * WAY + w], s);
  }
}

// ---------------- kernel 6: finalize logits ----------------
__global__ void finalize_kernel(const float* __restrict__ acc, float* __restrict__ out) {
  int i = threadIdx.x;
  if (i < NQ * WAY) out[i] = -acc[i] * (1.0f / (float)TT);
}

// ---------------- host launcher ----------------
extern "C" void kernel_launch(void* const* d_in, const int* in_sizes, int n_in,
                              void* d_out, int out_size, void* d_ws, size_t ws_size,
                              hipStream_t stream) {
  const float* support = (const float*)d_in[0];
  const float* queries = (const float*)d_in[1];
  // d_in[2] = support_labels (sorted, unused)
  const float* k_w = (const float*)d_in[3];
  const float* k_b = (const float*)d_in[4];
  const float* v_w = (const float*)d_in[5];
  const float* v_b = (const float*)d_in[6];
  const float* ln_g = (const float*)d_in[7];
  const float* ln_b = (const float*)d_in[8];

  char* ws = (char*)d_ws;
  bf16*  Xb   = (bf16*)(ws + OFF_XB);
  bf16*  Kq   = (bf16*)(ws + OFF_KQ);
  bf16*  Ks   = (bf16*)(ws + OFF_KS);
  bf16*  Vst  = (bf16*)(ws + OFF_VST);
  float* Acc  = (float*)(ws + OFF_ACC);
  bf16*  Wt   = (bf16*)(ws + OFF_WT);
  float* Aout = (float*)(ws + OFF_AOUT);
  float* Qv   = (float*)(ws + OFF_QV);

  zero_kernel<<<(ZERO_U4 + 255) / 256, 256, 0, stream>>>((uint4*)ws, ZERO_U4);
  prep_x_kernel<<<(NALL * SEQ * IND + 255) / 256, 256, 0, stream>>>(support, queries, Xb);
  prep_w_kernel<<<dim3(36, 36, 6), dim3(32, 8), 0, stream>>>(k_w, v_w, Wt);
  proj_kernel<<<dim3(NOUT / 64, MROWS_PAD / 64), 128, 0, stream>>>(Xb, Wt, Aout);
  gather_ln_kernel<<<dim3(NALL, TT), 128, 0, stream>>>(Aout, k_b, v_b, ln_g, ln_b,
                                                       Kq, Ks, Vst, Qv);
  attn_kernel<<<dim3(7, NQ, WAY * HEADS), 192, ATTN_LDS, stream>>>(Kq, Ks, Vst, Qv, Acc);
  finalize_kernel<<<1, 256, 0, stream>>>(Acc, (float*)d_out);
}
